// RNNRWW_29867202576576
// MI455X (gfx1250) — matrix operand order repaired and checked by reading, not compiled, not run on hardware
//
#include <hip/hip_runtime.h>

#define NREG 512
#define NS   64
#define TRS  20
#define STEPS 20
#define TSTEPS (TRS*STEPS)   // 400
#define SIM_BLOCKS 128       // 4 regions per block
#define RPB 4                // regions per block

// model constants
#define DT_     0.05f
#define W_E_    1.0f
#define W_I_    0.7f
#define TAU_E_  0.1f
#define TAU_I_  0.01f
#define GAM_E_  0.641f
#define GAM_I_  1.0f
#define I_0_    0.32f
#define I_EXT_  0.02f
#define A_Ec    310.0f
#define B_Ec    125.0f
#define D_Ec    0.16f
#define A_Ic    615.0f
#define B_Ic    177.0f
#define D_Ic    0.087f
#define ALPHA_  0.32f
#define RHO_    0.34f
#define K1_     2.38f
#define K2_     2.0f
#define K3_     0.48f
#define V0_     0.02f
#define E0_     0.34f
#define TAU_S_  0.65f
#define TAU_F_  0.41f
#define TAU_0_  0.98f

typedef __attribute__((ext_vector_type(16))) __bf16 v16bf;
typedef __attribute__((ext_vector_type(8)))  float  v8f;
typedef unsigned short u16;

union BF16x16 { u16 u[16]; uint4 q[2]; v16bf v; };

__device__ __forceinline__ float reluf(float x) { return fmaxf(x, 0.f); }

__device__ __forceinline__ u16 f2bf(float f) {
  unsigned u = __float_as_uint(f);
  u += 0x7fffu + ((u >> 16) & 1u);           // round-to-nearest-even
  return (u16)(u >> 16);
}

__device__ __forceinline__ float h_tf(float a, float b, float d, float z) {
  float x = a * z - b;
  return (1e-5f + fabsf(x)) / (1e-5f * d + fabsf(1.f - __expf(-d * x)));
}

__device__ __forceinline__ float wave_sum(float v) {
  #pragma unroll
  for (int o = 16; o > 0; o >>= 1) v += __shfl_xor(v, o, 32);
  return v;
}

// ---------------------------------------------------------------------------
// Prep 1: w = 0.5*(exp(gc)*sc + (exp(gc)*sc)^T), per-row sum-of-squares + row sum
// ---------------------------------------------------------------------------
__global__ __launch_bounds__(256) void k_prep_w(const float* __restrict__ sc,
                                                const float* __restrict__ gc,
                                                float* __restrict__ w,
                                                float* __restrict__ sqp,
                                                float* __restrict__ rsum) {
  const int i = blockIdx.x;
  float sq = 0.f, rs = 0.f;
  for (int j = threadIdx.x; j < NREG; j += 256) {
    float wv = 0.5f * (__expf(gc[i * NREG + j]) * sc[i * NREG + j] +
                       __expf(gc[j * NREG + i]) * sc[j * NREG + i]);
    w[i * NREG + j] = wv;
    sq += wv * wv;
    rs += wv;
  }
  sq = wave_sum(sq);
  rs = wave_sum(rs);
  __shared__ float sa[8], sb[8];
  if ((threadIdx.x & 31) == 0) { sa[threadIdx.x >> 5] = sq; sb[threadIdx.x >> 5] = rs; }
  __syncthreads();
  if (threadIdx.x == 0) {
    float ts = 0.f, tr2 = 0.f;
    #pragma unroll
    for (int k = 0; k < 8; ++k) { ts += sa[k]; tr2 += sb[k]; }
    sqp[i] = ts;
    rsum[i] = tr2;
  }
}

// ---------------------------------------------------------------------------
// Prep 2: rnorm = 1/||w||_F  (deterministic two-pass reduction)
// ---------------------------------------------------------------------------
__global__ __launch_bounds__(512) void k_prep_norm(const float* __restrict__ sqp,
                                                   float* __restrict__ rnorm) {
  float v = sqp[threadIdx.x];
  v = wave_sum(v);
  __shared__ float sa[16];
  if ((threadIdx.x & 31) == 0) sa[threadIdx.x >> 5] = v;
  __syncthreads();
  if (threadIdx.x == 0) {
    float t = 0.f;
    #pragma unroll
    for (int k = 0; k < 16; ++k) t += sa[k];
    rnorm[0] = 1.f / sqrtf(t);
  }
}

// ---------------------------------------------------------------------------
// Prep 3: lap = w*rn - diag(rowsum*rn), emitted in f32 (matvec) and bf16 (WMMA)
// ---------------------------------------------------------------------------
__global__ __launch_bounds__(256) void k_prep_lap(const float* __restrict__ w,
                                                  const float* __restrict__ rnorm,
                                                  const float* __restrict__ rsum,
                                                  float* __restrict__ lapf,
                                                  u16* __restrict__ lapbf) {
  const int i = blockIdx.x;
  const float rn = rnorm[0];
  const float dsub = rsum[i] * rn;
  for (int j = threadIdx.x; j < NREG; j += 256) {
    float l = w[i * NREG + j] * rn - (i == j ? dsub : 0.f);
    lapf[i * NREG + j] = l;
    lapbf[i * NREG + j] = f2bf(l);
  }
}

// ---------------------------------------------------------------------------
// Convert + transpose noise_E0 once: e0T[t][s][k] (bf16, k contiguous).
// Tile staged global->LDS with CDNA5 async DMA (ASYNCcnt-tracked), then
// transposed/converted out of LDS with coalesced packed stores.
// ---------------------------------------------------------------------------
__global__ __launch_bounds__(256) void k_cvt(const float* __restrict__ e0,
                                             u16* __restrict__ e0T) {
  const int t  = blockIdx.x >> 3;     // 0..399
  const int k0 = (blockIdx.x & 7) * 64;
  const int tid = threadIdx.x;
  __shared__ float tile[64][65];      // [k][s], padded

  #pragma unroll
  for (int p = 0; p < 4; ++p) {
    const int ki = p * 16 + (tid >> 4);
    const int si = (tid & 15) * 4;
    unsigned ldsoff = (unsigned)(size_t)&tile[ki][si];
    unsigned long long ga =
        (unsigned long long)(size_t)(e0 + ((size_t)t * NREG + k0 + ki) * NS + si);
    asm volatile("global_load_async_to_lds_b128 %0, %1, off"
                 :: "v"(ldsoff), "v"(ga) : "memory");
  }
  asm volatile("s_wait_asynccnt 0x0" ::: "memory");
  __syncthreads();

  unsigned* __restrict__ outw = (unsigned*)e0T;
  #pragma unroll
  for (int it = 0; it < 8; ++it) {
    const int lin = it * 256 + tid;   // 0..2047
    const int si2 = lin >> 5;         // 0..63
    const int k2  = (lin & 31) * 2;   // even k
    const unsigned pk = (unsigned)f2bf(tile[k2][si2]) |
                        ((unsigned)f2bf(tile[k2 + 1][si2]) << 16);
    outw[(((size_t)t * NS + si2) * NREG + k0 + k2) >> 1] = pk;
  }
}

// ---------------------------------------------------------------------------
// Batched GEMM: P[t] = lap(bf16) @ e0[t]  for all 400 micro-steps.
// One wave per 16x16 output tile; K=512 in 16 v_wmma_f32_16x16x32_bf16 ops.
// A layout (16-bit 16x32): lane = 16h + m; elems 0..7 -> K = kb+8h+e,
//                          elems 8..15 -> K = kb+16+8h+(e-8)   (ISA 7.12.2)
// B layout (16-bit 32x16): lane = 16h + n; same K mapping as A
//   -> with e0T[s][k] (k contiguous) this is two 16B contiguous loads.
// C/D (f32 16x16): elem j at lane 16h+n -> row m = 8h + j
// ---------------------------------------------------------------------------
__global__ __launch_bounds__(256) void k_gemm(const u16* __restrict__ lapbf,
                                              const u16* __restrict__ e0T,
                                              float* __restrict__ P) {
  const int lane = threadIdx.x & 31;
  const int wave = threadIdx.x >> 5;
  const int g = blockIdx.x * 8 + wave;   // tile id: mb fastest -> B-tile L2 reuse
  const int mb = g & 31;
  const int sb = (g >> 5) & 3;
  const int tt = g >> 7;                 // 0..399
  const int h = lane >> 4;
  const int l15 = lane & 15;
  const int m = mb * 16 + l15;
  const int scol = sb * 16 + l15;
  const u16* __restrict__ brow = e0T + ((size_t)tt * NS + scol) * NREG;
  const u16* __restrict__ arow = lapbf + (size_t)m * NREG + 8 * h;

  v8f acc = {};
  #pragma unroll 4
  for (int kb = 0; kb < NREG; kb += 32) {
    BF16x16 A;
    A.q[0] = *(const uint4*)(arow + kb);
    A.q[1] = *(const uint4*)(arow + kb + 16);
    BF16x16 Bv;
    const u16* bp = brow + kb + 8 * h;
    Bv.q[0] = *(const uint4*)(bp);
    Bv.q[1] = *(const uint4*)(bp + 16);
    acc = __builtin_amdgcn_wmma_f32_16x16x32_bf16(false, A.v, false, Bv.v,
                                                  (short)0, acc, false, false);
  }
  float* out = P + ((size_t)tt * NREG + mb * 16 + 8 * h) * NS + scol;
  #pragma unroll
  for (int j = 0; j < 8; ++j) out[(size_t)j * NS] = acc[j];
}

// ---------------------------------------------------------------------------
// Persistent simulation kernel: 128 blocks x 256 threads, global barrier/step.
// ---------------------------------------------------------------------------
__device__ __forceinline__ void gbar(unsigned* ctr, unsigned* barid) {
  __threadfence();          // release: my stores reach device scope
  __syncthreads();
  if (threadIdx.x == 0) {
    atomicAdd(ctr, 1u);
    const unsigned target = (*barid) * SIM_BLOCKS;
    while (__hip_atomic_load(ctr, __ATOMIC_RELAXED, __HIP_MEMORY_SCOPE_AGENT) < target)
      __builtin_amdgcn_s_sleep(2);
  }
  __syncthreads();
  __threadfence();          // acquire: invalidate stale WGP-local lines
  ++(*barid);
}

__global__ __launch_bounds__(256) void k_sim(
    const float* __restrict__ hx, const float* __restrict__ external,
    const float* __restrict__ nE0, const float* __restrict__ nI0,
    const float* __restrict__ nE, const float* __restrict__ nI,
    const float* __restrict__ nB, const float* __restrict__ lapf,
    const float* __restrict__ P, float* __restrict__ Em_g,
    const float* gS, const float* gEES, const float* gIES, const float* gEIS,
    const float* stdinS, const float* stdoutS,
    unsigned* ctr, float* __restrict__ out) {
  const int tid = threadIdx.x;
  const int r = tid >> 6;               // local region 0..3
  const int s = tid & 63;               // sample 0..63
  const int n = blockIdx.x * RPB + r;   // global region
  const int lane = tid & 31;

  __shared__ float lapRow[RPB][NREG];   // 8 KB, resident for all 400 steps
  __shared__ float EmS[NREG];
  __shared__ float mvred[8];
  __shared__ float redA[8], redB[8];
  __shared__ float EmL[RPB], ImL[RPB];

  const float gcpl = gS[0];
  const float g_EE_e = 0.001f + reluf(gEES[0]);
  const float g_IE_e = 0.001f + reluf(gIES[0]);
  const float g_EI_e = 0.001f + reluf(gEIS[0]);
  const float std_in_e = 0.02f + reluf(stdinS[0]);
  const float std_out_e = reluf(stdoutS[0]);
  const float sqdt = sqrtf(DT_);

  for (int j = tid; j < RPB * NREG; j += 256)
    lapRow[j >> 9][j & (NREG - 1)] =
        lapf[(blockIdx.x * RPB + (j >> 9)) * NREG + (j & (NREG - 1))];

  if (s == 0) {
    EmL[r] = hx[n * 6 + 0];
    ImL[r] = hx[n * 6 + 1];
    Em_g[n] = hx[n * 6 + 0];
  }
  float st_x = hx[n * 6 + 2], st_f = hx[n * 6 + 3];
  float st_v = hx[n * 6 + 4], st_q = hx[n * 6 + 5];

  // prefetch step-0 noise into registers (state-independent)
  size_t nbase = ((size_t)0 * NREG + n) * NS + s;
  float e0v = nE0[nbase], i0v = nI0[nbase], eEv = nE[nbase], eIv = nI[nbase];
  float Pv = P[nbase];

  unsigned barid = 1;
  gbar(ctr, &barid);                    // publish initial Em_g

  for (int tt = 0; tt < TSTEPS; ++tt) {
    const int trx = tt / STEPS;
    const int st = tt - trx * STEPS;

    // stage Em into LDS (2 elements / thread)
    EmS[tid] = Em_g[tid];
    EmS[tid + 256] = Em_g[tid + 256];
    __syncthreads();

    // matvec: lapEm[r] = lap[n,:] . Em
    float mv = 0.f;
    #pragma unroll
    for (int k = 0; k < 8; ++k) {
      int mm = s + 64 * k;
      mv += lapRow[r][mm] * EmS[mm];
    }
    mv = wave_sum(mv);
    if (lane == 0) mvred[tid >> 5] = mv;
    __syncthreads();
    const float lapEm = mvred[2 * r] + mvred[2 * r + 1];
    const float hxE = EmL[r], hxI = ImL[r];

    // elementwise micro-step
    const float ex = external[(n * STEPS + st) * TRS + trx];
    const float E = hxE + 0.02f * e0v;
    const float I = hxI + 0.001f * i0v;
    const float lapE = lapEm + 0.02f * Pv;
    const float IE = tanhf(reluf(W_E_ * I_0_ + I_EXT_ * ex + g_EE_e * E +
                                 gcpl * lapE - g_IE_e * I));
    const float II = tanhf(reluf(W_I_ * I_0_ + g_EI_e * E - I));
    const float rE = h_tf(A_Ec, B_Ec, D_Ec, IE);
    const float rI = h_tf(A_Ic, B_Ic, D_Ic, II);
    const float En = E + DT_ * (-E / TAU_E_ + GAM_E_ * (1.f - E) * rE) +
                     sqdt * std_in_e * eEv;
    const float In = I + DT_ * (-I / TAU_I_ + GAM_I_ * rI) +
                     sqdt * std_in_e * eIv;
    const float Ep = tanhf(reluf(En));
    const float Ip = tanhf(reluf(In));

    float se = wave_sum(Ep);
    float si = wave_sum(Ip);
    if (lane == 0) { redA[tid >> 5] = se; redB[tid >> 5] = si; }

    // prefetch next step's noise into registers before the barrier
    const int ttn = (tt < TSTEPS - 1) ? tt + 1 : tt;
    nbase = ((size_t)ttn * NREG + n) * NS + s;
    e0v = nE0[nbase]; i0v = nI0[nbase]; eEv = nE[nbase]; eIv = nI[nbase];
    Pv = P[nbase];
    if (tt < TSTEPS - 2) {              // hint prefetch two steps ahead
      size_t nb2 = ((size_t)(tt + 2) * NREG + n) * NS + s;
      __builtin_prefetch(&nE0[nb2], 0, 0);
      __builtin_prefetch(&nI0[nb2], 0, 0);
      __builtin_prefetch(&nE[nb2], 0, 0);
      __builtin_prefetch(&nI[nb2], 0, 0);
      __builtin_prefetch(&P[nb2], 0, 0);
    }
    __syncthreads();

    if (s == 0) {
      float Em = fmaxf((redA[2 * r] + redA[2 * r + 1]) * (1.f / 64.f), 1e-5f);
      float Im = fmaxf((redB[2 * r] + redB[2 * r + 1]) * (1.f / 64.f), 1e-5f);
      const float v_a = powf(st_v, 1.f / ALPHA_);
      const float xn = st_x + DT_ * (Em - st_x / TAU_S_ - (st_f - 1.f) / TAU_F_);
      const float fn = st_f + DT_ * st_x;
      const float vn = st_v + DT_ * (st_f - v_a) / TAU_0_;
      const float qn = st_q + DT_ * (st_f * (1.f - powf(1.f - RHO_, 1.f / st_f)) / RHO_ -
                                     st_q * v_a / st_v) / TAU_0_;
      st_x = tanhf(xn);
      st_f = 1.f + tanhf(fn - 1.f);
      st_v = 1.f + tanhf(vn - 1.f);
      st_q = 1.f + tanhf(qn - 1.f);
      EmL[r] = Em; ImL[r] = Im;
      Em_g[n] = Em;                     // published at the barrier
      if (st == STEPS - 1) {
        const float bold = std_out_e * nB[trx * NREG + n] +
            100.f * V0_ / E0_ * (K1_ * (1.f - st_q) +
                                 K2_ * (1.f - st_q / st_v) +
                                 K3_ * (1.f - st_v));
        out[n * TRS + trx] = bold;      // bolds.T -> (N, TRS)
      }
    }
    gbar(ctr, &barid);
  }
}

// ---------------------------------------------------------------------------
extern "C" void kernel_launch(void* const* d_in, const int* in_sizes, int n_in,
                              void* d_out, int out_size, void* d_ws, size_t ws_size,
                              hipStream_t stream) {
  const float* external = (const float*)d_in[0];
  const float* hx       = (const float*)d_in[1];
  // d_in[2] = hE (unused by the reference)
  const float* sc       = (const float*)d_in[3];
  const float* gc       = (const float*)d_in[4];
  const float* gS       = (const float*)d_in[5];
  const float* gEES     = (const float*)d_in[6];
  const float* gIES     = (const float*)d_in[7];
  const float* gEIS     = (const float*)d_in[8];
  const float* stdinS   = (const float*)d_in[9];
  const float* stdoutS  = (const float*)d_in[10];
  const float* nE0      = (const float*)d_in[11];
  const float* nI0      = (const float*)d_in[12];
  const float* nE       = (const float*)d_in[13];
  const float* nI       = (const float*)d_in[14];
  const float* nB       = (const float*)d_in[15];

  char* ws = (char*)d_ws;
  unsigned* ctr  = (unsigned*)(ws + 0);                       // 256 B
  float* rnorm   = (float*)(ws + 256);
  float* sqp     = (float*)(ws + 512);                        // 2 KB
  float* rsum    = (float*)(ws + 512 + 2048);                 // 2 KB
  float* w       = (float*)(ws + 8192);                       // 1 MB
  float* lapf    = (float*)(ws + 8192 + NREG * NREG * 4);     // 1 MB
  u16*   lapbf   = (u16*)(ws + 8192 + 2 * NREG * NREG * 4);   // 512 KB
  float* Em_g    = (float*)(ws + 8192 + 2 * NREG * NREG * 4 + NREG * NREG * 2);
  float* P       = (float*)(ws + 8192 + 2 * NREG * NREG * 4 + NREG * NREG * 2 + 4096);
  // P: 400*512*64*4 = 52.4 MB
  u16*   e0T     = (u16*)((char*)P + (size_t)TSTEPS * NREG * NS * 4);
  // e0T: 400*512*64*2 = 26.2 MB; total ws use ~81.3 MB

  hipMemsetAsync(ctr, 0, 256, stream);
  k_prep_w<<<NREG, 256, 0, stream>>>(sc, gc, w, sqp, rsum);
  k_prep_norm<<<1, 512, 0, stream>>>(sqp, rnorm);
  k_prep_lap<<<NREG, 256, 0, stream>>>(w, rnorm, rsum, lapf, lapbf);
  k_cvt<<<TSTEPS * 8, 256, 0, stream>>>(nE0, e0T);
  k_gemm<<<(TSTEPS * 32 * 4) / 8, 256, 0, stream>>>(lapbf, e0T, P);
  k_sim<<<SIM_BLOCKS, 256, 0, stream>>>(hx, external, nE0, nI0, nE, nI, nB,
                                        lapf, P, Em_g, gS, gEES, gIES, gEIS,
                                        stdinS, stdoutS, ctr, (float*)d_out);
}
